// TestConv3_18322330484758
// MI455X (gfx1250) — matrix operand order repaired and checked by reading, not compiled, not run on hardware
//
#include <hip/hip_runtime.h>

// ---------------------------------------------------------------------------
// CDNA5 / gfx1250 implementation. wave32. WMMA f16 (probe-confirmed builtin).
// ---------------------------------------------------------------------------

typedef __attribute__((ext_vector_type(16))) _Float16 v16h;
typedef __attribute__((ext_vector_type(8)))  _Float16 v8h;
typedef __attribute__((ext_vector_type(8)))  float    v8f;

#define SPLIT_C 812

// -------------------------- utility kernels --------------------------------

__global__ void k_zero_u32(unsigned* __restrict__ p, size_t nwords) {
    size_t i = (size_t)blockIdx.x * blockDim.x + threadIdx.x;
    size_t stride = (size_t)gridDim.x * blockDim.x;
    for (; i < nwords; i += stride) p[i] = 0u;
}

__global__ void k_cvt_f16(const float* __restrict__ src, _Float16* __restrict__ dst, int n) {
    int i = blockIdx.x * blockDim.x + threadIdx.x;
    if (i < n) dst[i] = (_Float16)src[i];
}

// -------------------------- edge phase -------------------------------------

__global__ void k_deg(const int* __restrict__ ecol, unsigned* __restrict__ deg, int E) {
    int e = blockIdx.x * blockDim.x + threadIdx.x;
    if (e < E) atomicAdd(&deg[ecol[e]], 1u);
}

__global__ void k_norm(const int* __restrict__ erow, const int* __restrict__ ecol,
                       const unsigned* __restrict__ deg, float* __restrict__ normv, int E) {
    int e = blockIdx.x * blockDim.x + threadIdx.x;
    if (e >= E) return;
    unsigned dr = deg[erow[e]];
    unsigned dc = deg[ecol[e]];
    float ir = dr ? rsqrtf((float)dr) : 0.0f;
    float ic = dc ? rsqrtf((float)dc) : 0.0f;
    normv[e] = ir * ic;
}

// one wave (32 lanes) per edge; lane covers 4 of 128 channels (float4)
__global__ __launch_bounds__(256) void k_scatter(
    const int* __restrict__ erow, const int* __restrict__ ecol,
    const float* __restrict__ normv, const float* __restrict__ x,
    float* __restrict__ x_same, float* __restrict__ x_diff, int E)
{
    long long gid = (long long)blockIdx.x * blockDim.x + threadIdx.x;
    int e    = (int)(gid >> 5);
    int lane = (int)(gid & 31);
    if (e >= E) return;
    float w = normv[e];
    if (w == 0.0f) return;
    int r = erow[e], c = ecol[e];
    bool same = (r < SPLIT_C) == (c < SPLIT_C);
    float* dst = (same ? x_same : x_diff) + (size_t)r * 128 + lane * 4;
    const float4 xv = *(const float4*)(x + (size_t)c * 128 + lane * 4);
    atomicAdd(dst + 0, w * xv.x);
    atomicAdd(dst + 1, w * xv.y);
    atomicAdd(dst + 2, w * xv.z);
    atomicAdd(dst + 3, w * xv.w);
}

// exact top-3 by (norm desc, edge idx asc) via 3-pass atomicMax cascade
__global__ void k_topk(const int* __restrict__ erow, const int* __restrict__ ecol,
                       const float* __restrict__ normv,
                       unsigned long long* __restrict__ s0,
                       unsigned long long* __restrict__ s1,
                       unsigned long long* __restrict__ s2,
                       int pass, int E)
{
    int e = blockIdx.x * blockDim.x + threadIdx.x;
    if (e >= E) return;
    int r = erow[e], c = ecol[e];
    if ((r < SPLIT_C) == (c < SPLIT_C)) return;  // only "diff" edges
    unsigned nb = __float_as_uint(normv[e]);     // norm >= 0 -> bits monotone
    unsigned long long key = ((unsigned long long)nb << 32) | (unsigned)(~(unsigned)e);
    if (pass == 0) {
        atomicMax(&s0[c], key);
    } else if (pass == 1) {
        if (key != s0[c]) atomicMax(&s1[c], key);
    } else {
        if (key != s0[c] && key != s1[c]) atomicMax(&s2[c], key);
    }
}

// bsum[t] = sum of x_dt over up-to-3 top neighbors; cnt[t] = #valid
__global__ void k_bsum(const float* __restrict__ xdt,
                       const unsigned long long* __restrict__ s0,
                       const unsigned long long* __restrict__ s1,
                       const unsigned long long* __restrict__ s2,
                       const int* __restrict__ erow,
                       float* __restrict__ bsum, unsigned* __restrict__ cnt, int total)
{
    int idx = blockIdx.x * blockDim.x + threadIdx.x;
    if (idx >= total) return;
    int t = idx >> 6, c = idx & 63;
    unsigned long long k0 = s0[t], k1 = s1[t], k2 = s2[t];
    float s = 0.0f; unsigned cn = 0;
    if (k0) { int e = (int)(~(unsigned)k0); s += xdt[(size_t)erow[e] * 64 + c]; ++cn; }
    if (k1) { int e = (int)(~(unsigned)k1); s += xdt[(size_t)erow[e] * 64 + c]; ++cn; }
    if (k2) { int e = (int)(~(unsigned)k2); s += xdt[(size_t)erow[e] * 64 + c]; ++cn; }
    bsum[idx] = s;
    if (c == 0) cnt[t] = cn;
}

// --------------------- dense GEMM: out = A(16n x 128) @ W^T + b ------------
// WMMA f16 16x16x32, f32 accumulate. One wave = 16 rows x 64 cols.
// A-frag layout (f16 16x32): lane m=lane&15; laneLo K={h,16+h}, laneHi +8.
// B-frag layout (f16 32x16): lane n=lane&15; laneLo K={h,16+h}, laneHi +8 (same as A).
__global__ __launch_bounds__(256) void k_gemm_nk128(
    const float* __restrict__ A, const float* __restrict__ W,
    const float* __restrict__ bias, float* __restrict__ out, int ntiles)
{
    int wave = threadIdx.x >> 5;
    int lane = threadIdx.x & 31;
    int tile = blockIdx.x * 8 + wave;
    if (tile >= ntiles) return;
    int row0 = tile * 16;
    int m  = lane & 15;
    int hi = lane >> 4;
    int n  = m;

    v8f acc[4] = {};
    for (int k0 = 0; k0 < 128; k0 += 32) {
        const float* ap = A + (size_t)(row0 + m) * 128 + k0 + (hi ? 8 : 0);
        v16h af;
        #pragma unroll
        for (int h = 0; h < 8; ++h) af[h] = (_Float16)ap[h];
        #pragma unroll
        for (int h = 0; h < 8; ++h) af[8 + h] = (_Float16)ap[16 + h];
        #pragma unroll
        for (int ot = 0; ot < 4; ++ot) {
            const float* wp = W + (size_t)(ot * 16 + n) * 128 + k0 + (hi ? 8 : 0);
            v16h bf;
            #pragma unroll
            for (int h = 0; h < 16; ++h) bf[h] = (_Float16)wp[h + (h & 8)];
            acc[ot] = __builtin_amdgcn_wmma_f32_16x16x32_f16(
                false, af, false, bf, (short)0, acc[ot], false, false);
        }
    }
    #pragma unroll
    for (int ot = 0; ot < 4; ++ot) {
        float bv = bias[ot * 16 + n];
        #pragma unroll
        for (int g = 0; g < 8; ++g) {
            int r = row0 + g + (hi ? 8 : 0);   // C layout: M = g + 8*hi, N = lane&15
            out[(size_t)r * 64 + ot * 16 + n] = acc[ot][g] + bv;
        }
    }
}

// --------------------- bilinear via on-the-fly outer products ---------------
// y[t,o] = sum_k a[t,k/64]*bsum[t,k%64] * Wh[o*4096+k],  K = 4096
// then: out[t,o] += gate*lrelu(y/cnt + b_bil) + (1-gate)*x_dt   (cnt>0)
//       out[t,o] += (1-gate)*x_dt                               (cnt==0)
__global__ __launch_bounds__(256) void k_bilinear(
    const float* __restrict__ xdt, const float* __restrict__ bsum,
    const _Float16* __restrict__ Wh, const unsigned* __restrict__ cnt,
    const float* __restrict__ b_bil, const float* __restrict__ gate_w,
    float* __restrict__ out, int ntiles)
{
    __shared__ __align__(16) _Float16 smem[8 * 2 * 16 * 72]; // 8 waves x (a,b) x 16x72 (padded)
    int wave = threadIdx.x >> 5;
    int lane = threadIdx.x & 31;
    int tile = blockIdx.x * 8 + wave;
    bool active = tile < ntiles;
    _Float16* a_s = smem + wave * (2 * 16 * 72);
    _Float16* b_s = a_s + 16 * 72;
    int t0 = tile * 16;

    if (active) {  // stage f32->f16 tiles into this wave's LDS slice
        int r = lane & 15;
        int half = lane >> 4;
        const float* xa = xdt  + (size_t)(t0 + r) * 64 + half * 32;
        const float* xb = bsum + (size_t)(t0 + r) * 64 + half * 32;
        #pragma unroll
        for (int c = 0; c < 32; ++c) {
            a_s[r * 72 + half * 32 + c] = (_Float16)xa[c];
            b_s[r * 72 + half * 32 + c] = (_Float16)xb[c];
        }
    }
    __syncthreads();

    if (active) {
        int m  = lane & 15;
        int hi = lane >> 4;
        int n  = m;
        v8f acc[4] = {};
        for (int k0 = 0; k0 < 4096; k0 += 32) {
            int i0 = k0 >> 6;        // a-channel for this K block
            int j0 = k0 & 63;        // 0 or 32: bsum-channel base
            _Float16 aval = a_s[m * 72 + i0];
            const v8h* bp = (const v8h*)&b_s[m * 72 + j0 + (hi ? 8 : 0)];
            v8h blo = bp[0];         // j = j0 + 8*hi + h       (h = 0..7)
            v8h bhi = bp[2];         // j = j0 + 8*hi + 16 + h  (h = 8..15)
            v16h af;
            #pragma unroll
            for (int h = 0; h < 8; ++h) af[h] = aval * blo[h];
            #pragma unroll
            for (int h = 0; h < 8; ++h) af[8 + h] = aval * bhi[h];
            #pragma unroll
            for (int ot = 0; ot < 4; ++ot) {
                const v8h* wv = (const v8h*)(Wh + (size_t)(ot * 16 + n) * 4096
                                             + k0 + (hi ? 8 : 0));
                v8h w0 = wv[0], w1 = wv[2];
                v16h bf;
                #pragma unroll
                for (int h = 0; h < 8; ++h) { bf[h] = w0[h]; bf[8 + h] = w1[h]; }
                acc[ot] = __builtin_amdgcn_wmma_f32_16x16x32_f16(
                    false, af, false, bf, (short)0, acc[ot], false, false);
            }
        }
        // fused epilogue
        #pragma unroll
        for (int ot = 0; ot < 4; ++ot) {
            int o = ot * 16 + n;
            float bb = b_bil[o];
            float g  = 1.0f / (1.0f + __expf(-gate_w[o]));
            #pragma unroll
            for (int gg = 0; gg < 8; ++gg) {
                int t = t0 + gg + (hi ? 8 : 0);
                unsigned cn = cnt[t];
                float bfv = (cn > 0) ? (acc[ot][gg] / (float)cn + bb) : 0.0f;
                float lr  = bfv > 0.0f ? bfv : 0.01f * bfv;
                float xd  = xdt[(size_t)t * 64 + o];
                out[(size_t)t * 64 + o] += g * lr + (1.0f - g) * xd;
            }
        }
    }
}

// ---------------------------------------------------------------------------

extern "C" void kernel_launch(void* const* d_in, const int* in_sizes, int n_in,
                              void* d_out, int out_size, void* d_ws, size_t ws_size,
                              hipStream_t stream)
{
    (void)n_in; (void)out_size; (void)ws_size;
    const float* x      = (const float*)d_in[0];
    const int*   erow   = (const int*)d_in[1];
    const float* W_same = (const float*)d_in[2];
    const float* b_same = (const float*)d_in[3];
    const float* W_diff = (const float*)d_in[4];
    const float* b_diff = (const float*)d_in[5];
    const float* W_bil  = (const float*)d_in[6];
    const float* b_bil  = (const float*)d_in[7];
    const float* gate_w = (const float*)d_in[8];

    const int N = in_sizes[0] / 128;
    const int E = in_sizes[1] / 2;
    const int* ecol = erow + E;

    // ---- workspace carve-up (256B aligned slices) ----
    char* wsb = (char*)d_ws;
    size_t off = 0;
    auto take = [&](size_t bytes) -> char* {
        char* p = wsb + off;
        off = (off + bytes + 255) & ~(size_t)255;
        return p;
    };
    unsigned*           deg    = (unsigned*)           take((size_t)N * 4);
    unsigned long long* s0     = (unsigned long long*) take((size_t)N * 8);
    unsigned long long* s1     = (unsigned long long*) take((size_t)N * 8);
    unsigned long long* s2     = (unsigned long long*) take((size_t)N * 8);
    float*              x_same = (float*)              take((size_t)N * 128 * 4);
    float*              x_diff = (float*)              take((size_t)N * 128 * 4);
    size_t zero_bytes = off;   // deg + slots + accumulators must start at 0
    float*              normv  = (float*)              take((size_t)E * 4);
    float*              x_dt   = (float*)              take((size_t)N * 64 * 4);
    float*              bsum   = (float*)              take((size_t)N * 64 * 4);
    unsigned*           cnt    = (unsigned*)           take((size_t)N * 4);
    _Float16*           Wh     = (_Float16*)           take((size_t)64 * 4096 * 2);
    float*              out    = (float*)d_out;

    // 1) zero the accumulator region
    k_zero_u32<<<1024, 256, 0, stream>>>((unsigned*)wsb, zero_bytes / 4);
    // 2) W_bil -> f16
    k_cvt_f16<<<(64 * 4096 + 255) / 256, 256, 0, stream>>>(W_bil, Wh, 64 * 4096);
    // 3) degrees
    k_deg<<<(E + 255) / 256, 256, 0, stream>>>(ecol, deg, E);
    // 4) per-edge symmetric norm
    k_norm<<<(E + 255) / 256, 256, 0, stream>>>(erow, ecol, deg, normv, E);
    // 5) masked scatter-sums (one wave per edge)
    {
        long long threads = (long long)E * 32;
        int blocks = (int)((threads + 255) / 256);
        k_scatter<<<blocks, 256, 0, stream>>>(erow, ecol, normv, x, x_same, x_diff, E);
    }
    // 6) exact top-3 per destination node (3-pass atomicMax cascade)
    k_topk<<<(E + 255) / 256, 256, 0, stream>>>(erow, ecol, normv, s0, s1, s2, 0, E);
    k_topk<<<(E + 255) / 256, 256, 0, stream>>>(erow, ecol, normv, s0, s1, s2, 1, E);
    k_topk<<<(E + 255) / 256, 256, 0, stream>>>(erow, ecol, normv, s0, s1, s2, 2, E);
    // 7) dense projections via WMMA:  x_st -> out,  x_dt -> ws
    int ntiles = N / 16;   // N = 20000 is a multiple of 16
    k_gemm_nk128<<<(ntiles + 7) / 8, 256, 0, stream>>>(x_same, W_same, b_same, out, ntiles);
    k_gemm_nk128<<<(ntiles + 7) / 8, 256, 0, stream>>>(x_diff, W_diff, b_diff, x_dt, ntiles);
    // 8) neighbor pre-sum (bilinearity: sum neighbors before the bilinear form)
    k_bsum<<<(N * 64 + 255) / 256, 256, 0, stream>>>(x_dt, s0, s1, s2, erow, bsum, cnt, N * 64);
    // 9) bilinear WMMA (K=4096 outer-product GEMM) + fused gate/leaky-relu epilogue
    k_bilinear<<<(ntiles + 7) / 8, 256, 0, stream>>>(x_dt, bsum, Wh, cnt, b_bil, gate_w, out, ntiles);
}
